// Function_22780506538393
// MI455X (gfx1250) — matrix operand order repaired and checked by reading, not compiled, not run on hardware
//
#include <hip/hip_runtime.h>
#include <hip/hip_bf16.h>

typedef __attribute__((ext_vector_type(16))) _Float16 v16h;
typedef __attribute__((ext_vector_type(8)))  _Float16 v8h;
typedef __attribute__((ext_vector_type(8)))  float    v8f;

// Problem constants (match reference)
#define BATCH   65536
#define IN_F    128
#define OUT_F   16
#define NBASIS  16
#define KDIM    (IN_F * NBASIS)   // 2048
#define CHUNKS  (KDIM / 32)       // 64 WMMA K-chunks of 32

// ---------------------------------------------------------------------------
// K-dimension permutation (free choice, applied identically to A and B):
//   chunk c, K_local 0..15  -> feature c,     basis K_local
//   chunk c, K_local 16..31 -> feature c+64,  basis K_local-16
// This makes each lane's feature index walk stride-1 across chunks
// (feature = c + 64*(lane>=16)), enabling float4 x-loads covering 4 chunks.
// ---------------------------------------------------------------------------

// Pre-kernel: w [16][128][16] f32 -> f16, pre-swizzled into the exact
// V_WMMA_F32_16X16X32_F16 A-fragment layout (A = weights, M = out-feature).
// A-fragment (16-bit A 16x32, wave32): lane l holds M = l&15;
//   element jj (0..15): K_local = 16*(l>>4) + jj (contiguous k order, same as B).
// Stored as 16-byte granules: ws[((c*2 + h)*32 + l)*8 + j], h = jj>>3.
__global__ __launch_bounds__(256) void prep_w_kernel(const float* __restrict__ w,
                                                     _Float16* __restrict__ wsA) {
    int idx = blockIdx.x * 256 + threadIdx.x;       // 0 .. 32767
    int c   = idx >> 9;                             // chunk
    int r   = idx & 511;
    int h   = r >> 8;                               // granule half (0/1)
    int l   = (r >> 3) & 31;                        // lane
    int j   = r & 7;                                // element within granule
    int m   = l & 15;                               // out-feature (M)
    int Kl  = ((l >> 4) << 4) + (h << 3) + j;       // K_local in chunk
    int i   = (Kl < 16) ? c : (c + 64);             // input feature (permuted K)
    int n   = Kl & 15;                              // basis order
    wsA[idx] = (_Float16)w[m * KDIM + i * NBASIS + n];
}

// Chebyshev basis T_0..T_15 of one x value, packed as a B-fragment lane slice.
__device__ __forceinline__ v16h cheb16(float xv) {
    v16h b;
    b[0] = (_Float16)1.0f;
    b[1] = (_Float16)xv;
    float t0 = 1.0f, t1 = xv;
    const float twox = xv + xv;
    #pragma unroll
    for (int n = 2; n < NBASIS; ++n) {
        float t2 = fmaf(twox, t1, -t0);             // T_n = 2x*T_{n-1} - T_{n-2}
        b[n] = (_Float16)t2;
        t0 = t1; t1 = t2;
    }
    return b;
}

// ---------------------------------------------------------------------------
// Main kernel: fused Chebyshev basis + GEMM via V_WMMA_F32_16X16X32_F16.
// Block = 256 threads = 8 waves; each wave computes a 16(batch) x 16(out) tile
// with the full K=2048 reduction (64 WMMAs), 4 chunks per loop body.
// ---------------------------------------------------------------------------
__global__ __launch_bounds__(256) void kan_cheb_wmma_kernel(const float* __restrict__ x,
                                                            const _Float16* __restrict__ wsA,
                                                            float* __restrict__ out) {
    // 64 KB of LDS: the whole pre-swizzled f16 weight matrix, as 16B granules.
    __shared__ v8h ldsW[CHUNKS * 64];               // 4096 * 16B = 65536 B

    const int t = threadIdx.x;

    // Cooperative, fully coalesced copy of the 64KB weight image into LDS.
    {
        const uint4* src = (const uint4*)wsA;
        uint4*       dst = (uint4*)ldsW;
        #pragma unroll
        for (int k = 0; k < 16; ++k)
            dst[t + k * 256] = src[t + k * 256];
    }
    __syncthreads();

    const int wave = t >> 5;
    const int lane = t & 31;
    const int hi   = lane >> 4;                     // wave half
    const int tile = blockIdx.x * 8 + wave;         // 16-row batch tile
    const int row  = tile * 16 + (lane & 15);       // this lane's batch row

    // Lane's feature index = c + 64*hi  ->  stride-1 walk, float4 every 4 chunks.
    const float* __restrict__ xrow = x + (size_t)row * IN_F + hi * 64;

    v8f acc = {};                                   // 16x16 f32 C/D tile

    for (int c4 = 0; c4 < CHUNKS / 4; ++c4) {
        const int c = c4 * 4;

        // One b128 x-load feeds 4 chunks (features c..c+3 for this lane half).
        const float4 xq = *(const float4*)(xrow + c);

        // Four independent B fragments (kept simultaneously live so the
        // register allocator cannot recycle B regs across the WMMA burst).
        v16h b0 = cheb16(xq.x);
        v16h b1 = cheb16(xq.y);
        v16h b2 = cheb16(xq.z);
        v16h b3 = cheb16(xq.w);

        // Four A fragments from LDS (2x ds_load_b128 each, conflict-free).
        union { v16h v; v8h h8[2]; } a0, a1, a2, a3;
        a0.h8[0] = ldsW[(c + 0) * 64 + lane]; a0.h8[1] = ldsW[(c + 0) * 64 + 32 + lane];
        a1.h8[0] = ldsW[(c + 1) * 64 + lane]; a1.h8[1] = ldsW[(c + 1) * 64 + 32 + lane];
        a2.h8[0] = ldsW[(c + 2) * 64 + lane]; a2.h8[1] = ldsW[(c + 2) * 64 + 32 + lane];
        a3.h8[0] = ldsW[(c + 3) * 64 + lane]; a3.h8[1] = ldsW[(c + 3) * 64 + 32 + lane];

        // Back-to-back WMMA burst: D->C accumulation chains at full rate
        // (no D->A/B hazard between these four).
        acc = __builtin_amdgcn_wmma_f32_16x16x32_f16(false, a0.v, false, b0, (short)0, acc, false, false);
        acc = __builtin_amdgcn_wmma_f32_16x16x32_f16(false, a1.v, false, b1, (short)0, acc, false, false);
        acc = __builtin_amdgcn_wmma_f32_16x16x32_f16(false, a2.v, false, b2, (short)0, acc, false, false);
        acc = __builtin_amdgcn_wmma_f32_16x16x32_f16(false, a3.v, false, b3, (short)0, acc, false, false);
    }

    // D element r: M(out) = r + 8*hi, N(batch) = lane&15.
    // out is [B,16] row-major -> each lane stores 8 consecutive f32 (32B aligned).
    float* outp = out + (size_t)row * OUT_F + hi * 8;
    *(v8f*)outp = acc;
}

extern "C" void kernel_launch(void* const* d_in, const int* in_sizes, int n_in,
                              void* d_out, int out_size, void* d_ws, size_t ws_size,
                              hipStream_t stream) {
    const float* x = (const float*)d_in[0];         // [65536, 128] f32
    const float* w = (const float*)d_in[1];         // [16, 128, 16] f32
    float*     out = (float*)d_out;                 // [65536, 16] f32
    _Float16*  wsA = (_Float16*)d_ws;               // 64 KB swizzled f16 weights

    // 1) Swizzle+convert weights into WMMA A-fragment order (32768 elems).
    prep_w_kernel<<<128, 256, 0, stream>>>(w, wsA);

    // 2) Fused basis + GEMM: 512 blocks x 8 waves x 16 rows = 65536 rows.
    kan_cheb_wmma_kernel<<<BATCH / 128, 256, 0, stream>>>(x, wsA, out);
}
